// SparseConvBlock_77902116815210
// MI455X (gfx1250) — matrix operand order repaired and checked by reading, not compiled, not run on hardware
//
#include <hip/hip_runtime.h>

typedef __bf16 v16bf __attribute__((ext_vector_type(16)));
typedef __bf16 v8bf  __attribute__((ext_vector_type(8)));
typedef float  v8f   __attribute__((ext_vector_type(8)));
typedef int    v4i   __attribute__((ext_vector_type(4)));

#if __has_builtin(__builtin_amdgcn_global_load_async_to_lds_b128) && \
    __has_builtin(__builtin_amdgcn_s_wait_asynccnt)
#define USE_ASYNC_LDS 1
typedef __attribute__((address_space(1))) v4i* gv4i_p;   // global
typedef __attribute__((address_space(3))) v4i* lv4i_p;   // LDS
#else
#define USE_ASYNC_LDS 0
#endif

// ---------------------------------------------------------------------------
// Kernel 1: convert features f32 -> bf16 (halves gather bandwidth in conv)
// ---------------------------------------------------------------------------
__global__ void cvt_feats_kernel(const float* __restrict__ f,
                                 __bf16* __restrict__ fb, int total) {
  int i = blockIdx.x * blockDim.x + threadIdx.x;
  if (i < total) fb[i] = (__bf16)f[i];
}

// ---------------------------------------------------------------------------
// Kernel 2: W[o][ic][oc] f32 -> Wt[o][oc][ic] bf16 (B-fragment reads become
// contiguous 16B chunks per lane)
// ---------------------------------------------------------------------------
__global__ void cvt_w_kernel(const float* __restrict__ W,
                             __bf16* __restrict__ Wt, int total) {
  int i = blockIdx.x * blockDim.x + threadIdx.x;
  if (i < total) {
    int o   = i >> 12;        // / (64*64)
    int rem = i & 4095;
    int oc  = rem >> 6;
    int ic  = rem & 63;
    Wt[i] = (__bf16)W[(o << 12) + (ic << 6) + oc];
  }
}

__device__ __forceinline__ int lower_bound_u(const int* __restrict__ a, int n,
                                             int key) {
  int lo = 0, hi = n;
  while (lo < hi) {
    int mid = (lo + hi) >> 1;
    if (a[mid] < key) lo = mid + 1; else hi = mid;
  }
  return lo;
}

// ---------------------------------------------------------------------------
// Kernel 3: sparse conv, gather -> WMMA(bf16, f32 acc) -> direct tile store.
// Block = 256 threads (8 waves). Block handles 64 output rows x 64 out chans.
// Wave w: out-channel group g = w&3, m-subtiles (w>>2) and (w>>2)+2, sharing
// one B fragment per (offset, kb). Per offset: two binary searches bound the
// compacted kernel-map segment hitting [base, base+64); those <=64 feature
// rows are gathered (async-to-LDS if available) and consumed by
// 2(kb) x 2(subtile) v_wmma_f32_16x16x32_bf16 per wave.
// ---------------------------------------------------------------------------
__global__ __launch_bounds__(256) void spconv_wmma_kernel(
    const __bf16* __restrict__ featsB, const __bf16* __restrict__ Wt,
    const int* __restrict__ in_idx, const int* __restrict__ out_idx,
    float* __restrict__ out, int N, int KOFF) {
  // 72-element row stride: 144B = 36 banks -> the 16 rows read by one wave's
  // A-fragment hit 16 disjoint 4-bank windows (conflict-free).
  __shared__ __bf16 As[64][72];    // 9216 B

  const int tid  = threadIdx.x;
  const int lane = tid & 31;
  const int w    = tid >> 5;       // wave id 0..7
  const int g    = w & 3;          // out-channel group (16 channels)
  const int wm   = w >> 2;         // wave handles m-subtiles wm and wm+2
  const int base = blockIdx.x * 64;
  const int n    = lane & 15;      // A row (M) / B column / C column
  const int kh   = lane >> 4;      // lane half selects K sub-range

  v8f acc[2];
#pragma unroll
  for (int s = 0; s < 2; ++s)
#pragma unroll
    for (int r = 0; r < 8; ++r) acc[s][r] = 0.f;

  const int chunk = tid & 7;       // 16B chunk within a 128B feature row
  const int rhalf = tid >> 3;      // 0..31: row (zero) / entry (fill) index

  for (int o = 0; o < KOFF; ++o) {
    const int* oi = out_idx + (size_t)o * N;
    const int* ii = in_idx  + (size_t)o * N;

    // bound the compacted segment whose outputs land in [base, base+64);
    // clamp the upper key to N so sentinel padding (== N) is excluded
    int lo   = lower_bound_u(oi, N, base);
    int keyh = base + 64 < N ? base + 64 : N;
    int hi   = lower_bound_u(oi, N, keyh);

    // zero the A tile (rows with no contributing entry stay zero)
    {
      v8bf z;
#pragma unroll
      for (int e = 0; e < 8; ++e) z[e] = (__bf16)0.f;
      *(v8bf*)&As[rhalf][chunk * 8]      = z;
      *(v8bf*)&As[rhalf + 32][chunk * 8] = z;
    }
    __syncthreads();

    // gather valid feature rows: up to 64 entries x 8 chunks of 16B
#pragma unroll
    for (int pass = 0; pass < 2; ++pass) {
      int e = rhalf + pass * 32;
      if (lo + e < hi) {
        int p   = lo + e;
        int row = oi[p] - base;                     // 0..63
        int src = ii[p];
        const __bf16* gsrc = featsB + ((size_t)src << 6) + chunk * 8;
        __bf16* ldst       = &As[row][chunk * 8];
#if USE_ASYNC_LDS
        __builtin_amdgcn_global_load_async_to_lds_b128(
            (gv4i_p)gsrc, (lv4i_p)ldst, 0, 0);
#else
        *(v8bf*)ldst = *(const v8bf*)gsrc;
#endif
      }
    }
#if USE_ASYNC_LDS
    __builtin_amdgcn_s_wait_asynccnt(0);
#endif
    __syncthreads();

    // D += A(16x64) x W_o(64x16): per wave 2 kb-steps x 2 m-subtiles,
    // B fragment shared by both subtiles
#pragma unroll
    for (int kb = 0; kb < 2; ++kb) {
      // B frag: lane column = 16g+n; K = kb*32 + kh*16 + 0..15 (contiguous)
      const __bf16* bp =
          Wt + (((size_t)o * 64 + g * 16 + n) << 6) + kb * 32 + kh * 16;
      v16bf b;
#pragma unroll
      for (int e = 0; e < 16; ++e) b[e] = bp[e];

#pragma unroll
      for (int s = 0; s < 2; ++s) {
        int ms = wm + 2 * s;
        // A frag: lane row = ms*16 + n; elems 0..15 -> K = kb*32+kh*16+0..15
        const __bf16* ap = &As[ms * 16 + n][kb * 32 + kh * 16];
        v16bf a;
#pragma unroll
        for (int e = 0; e < 8; ++e) { a[e] = ap[e]; a[e + 8] = ap[e + 8]; }

        acc[s] = __builtin_amdgcn_wmma_f32_16x16x32_bf16(
            false, a, false, b, (short)0, acc[s], false, false);
      }
    }
    __syncthreads();
  }

  // C layout: VGPR r -> M = r + 8*kh, column = lane&15
#pragma unroll
  for (int s = 0; s < 2; ++s) {
    int ms = wm + 2 * s;
#pragma unroll
    for (int r = 0; r < 8; ++r) {
      int row = base + ms * 16 + r + 8 * kh;
      if (row < N) out[((size_t)row << 6) + g * 16 + n] = acc[s][r];
    }
  }
}

// ---------------------------------------------------------------------------
// Kernel 4: deterministic per-channel partial sum / sumsq over conv output.
// Fixed grid (NB blocks), fixed strided order -> bitwise deterministic.
// ---------------------------------------------------------------------------
__global__ __launch_bounds__(256) void bn_partial_kernel(
    const float* __restrict__ out, float* __restrict__ partial, int N, int nblk) {
  __shared__ float redS[256], redQ[256];
  int t = threadIdx.x;
  int c = t & 63, rl = t >> 6;
  float s = 0.f, q = 0.f;
  for (int row = blockIdx.x * 4 + rl; row < N; row += nblk * 4) {
    float v = out[((size_t)row << 6) + c];
    s += v; q += v * v;
  }
  redS[t] = s; redQ[t] = q;
  __syncthreads();
  if (rl == 0) {
    float ss = redS[c] + redS[64 + c] + redS[128 + c] + redS[192 + c];
    float qq = redQ[c] + redQ[64 + c] + redQ[128 + c] + redQ[192 + c];
    partial[blockIdx.x * 64 + c] = ss;
    partial[(size_t)nblk * 64 + blockIdx.x * 64 + c] = qq;
  }
}

// ---------------------------------------------------------------------------
// Kernel 5: fold partials -> per-channel scale/bias (BN train-mode, biased var)
// ---------------------------------------------------------------------------
__global__ void bn_finalize_kernel(const float* __restrict__ partial,
                                   const float* __restrict__ gamma,
                                   const float* __restrict__ beta,
                                   float* __restrict__ sb, int N, int nblk) {
  int c = threadIdx.x;
  if (c < 64) {
    float s = 0.f, q = 0.f;
    for (int b = 0; b < nblk; ++b) {
      s += partial[b * 64 + c];
      q += partial[(size_t)nblk * 64 + b * 64 + c];
    }
    float mean = s / (float)N;
    float var  = q / (float)N - mean * mean;
    float inv  = rsqrtf(var + 1e-5f);
    float sc   = gamma[c] * inv;
    sb[c]      = sc;
    sb[64 + c] = beta[c] - mean * sc;
  }
}

// ---------------------------------------------------------------------------
// Kernel 6: apply BN + ReLU in place on d_out
// ---------------------------------------------------------------------------
__global__ void bn_apply_kernel(float* __restrict__ out,
                                const float* __restrict__ sb, int total) {
  int i = blockIdx.x * blockDim.x + threadIdx.x;
  if (i < total) {
    int c = i & 63;
    float v = out[i] * sb[c] + sb[64 + c];
    out[i] = v > 0.f ? v : 0.f;
  }
}

// ---------------------------------------------------------------------------
extern "C" void kernel_launch(void* const* d_in, const int* in_sizes, int n_in,
                              void* d_out, int out_size, void* d_ws, size_t ws_size,
                              hipStream_t stream) {
  (void)n_in; (void)out_size; (void)ws_size;
  const float* feats   = (const float*)d_in[0];
  const float* W       = (const float*)d_in[1];
  const float* gamma   = (const float*)d_in[2];
  const float* beta    = (const float*)d_in[3];
  const int*   in_idx  = (const int*)d_in[4];
  const int*   out_idx = (const int*)d_in[5];

  const int N    = in_sizes[0] / 64;        // active voxels
  const int KOFF = in_sizes[4] / N;         // 27 kernel offsets
  float* out = (float*)d_out;

  // workspace layout (all 256B aligned):
  //   featsB : N*64 bf16
  //   Wt     : KOFF*64*64 bf16 (transposed)
  //   partial: NB*64*2 f32
  //   sb     : 128 f32 (scale, bias)
  char* ws = (char*)d_ws;
  const int NB = 256;
  size_t featsBytes = (((size_t)N * 64 * 2) + 255) & ~(size_t)255;
  size_t wtBytes    = (((size_t)KOFF * 64 * 64 * 2) + 255) & ~(size_t)255;
  __bf16* featsB  = (__bf16*)ws;
  __bf16* Wt      = (__bf16*)(ws + featsBytes);
  float*  partial = (float*)(ws + featsBytes + wtBytes);
  float*  sb      = (float*)(ws + featsBytes + wtBytes + (size_t)NB * 64 * 2 * 4);

  int totF = N * 64;
  int totW = KOFF * 64 * 64;

  cvt_feats_kernel<<<(totF + 255) / 256, 256, 0, stream>>>(feats, featsB, totF);
  cvt_w_kernel<<<(totW + 255) / 256, 256, 0, stream>>>(W, Wt, totW);
  spconv_wmma_kernel<<<(N + 63) / 64, 256, 0, stream>>>(
      featsB, Wt, in_idx, out_idx, out, N, KOFF);
  bn_partial_kernel<<<NB, 256, 0, stream>>>(out, partial, N, NB);
  bn_finalize_kernel<<<1, 64, 0, stream>>>(partial, gamma, beta, sb, N, NB);
  bn_apply_kernel<<<(totF + 255) / 256, 256, 0, stream>>>(out, sb, totF);
}
